// MyBiRNN_73323681677974
// MI455X (gfx1250) — matrix operand order repaired and checked by reading, not compiled, not run on hardware
//
#include <hip/hip_runtime.h>
#include <cmath>

// Problem constants (from reference)
#define T_LEN 2048
#define B_SZ  128
#define I_SZ  128
#define H_SZ  256
#define NKB   12      // (I+H)/32 k-blocks of 32 for bf16 WMMA 16x16x32

typedef __attribute__((ext_vector_type(16))) __bf16 v16bf;
typedef __attribute__((ext_vector_type(8)))  float  v8f;

// Branch-free tanh: native v_tanh_f32 on gfx1250.
__device__ __forceinline__ float fast_tanh(float x) {
#if __has_builtin(__builtin_amdgcn_tanhf)
  return __builtin_amdgcn_tanhf(x);
#elif __has_builtin(__builtin_amdgcn_tanh_f32)
  return __builtin_amdgcn_tanh_f32(x);
#else
  // tanh(x) = 1 - 2/(exp(2x)+1);  2*log2(e) = 2.8853900817779268
  float e = __builtin_amdgcn_exp2f(x * 2.8853900817779268f);
  return 1.0f - 2.0f * __builtin_amdgcn_rcpf(e + 1.0f);
#endif
}

// A-fragment (16x32 bf16) per-lane layout: lanes 0-15 hold K {0..7,16..23},
// lanes 16-31 hold K {8..15,24..31} of their row. Two 16B LDS reads.
__device__ __forceinline__ v16bf load_a_frag(const __bf16* rowbase, int off) {
  union { uint4 u[2]; v16bf v; } r;
  r.u[0] = *(const uint4*)(rowbase + off);
  r.u[1] = *(const uint4*)(rowbase + off + 16);
  return r.v;
}

// A-fragment source selection for combined K = [x (4 blocks) | h (8 blocks)]
__device__ __forceinline__ const __bf16* a_src(int kb, const __bf16* xrb,
                                               const __bf16* hrb) {
  return (kb < 4) ? xrb : hrb;
}
__device__ __forceinline__ int a_off(int kb, int asel) {
  return ((kb < 4) ? kb : (kb - 4)) * 32 + asel;
}

__device__ __forceinline__ unsigned pack2_bf16(float a, float b) {
  union { __bf16 h[2]; unsigned u; } p;
  p.h[0] = (__bf16)a; p.h[1] = (__bf16)b;
  return p.u;
}

// ---------------------------------------------------------------------------
// Persistent scan kernel: one block = (direction, 16-row batch tile).
// 512 threads = 16 wave32; wave w computes the 16x16 tile of h columns
// [16w, 16w+16). Weights held in registers as WMMA B-fragments.
// ---------------------------------------------------------------------------
__global__ __launch_bounds__(512, 1) void birnn_scan(
    const float* __restrict__ x,                       // [B, T, I]
    const float* __restrict__ Wxh_f, const float* __restrict__ Whh_f,
    const float* __restrict__ bh_f,
    const float* __restrict__ Wxh_b, const float* __restrict__ Whh_b,
    const float* __restrict__ bh_b,
    float* __restrict__ hcat)                          // [B, 2H]
{
  __shared__ __align__(16) __bf16 hls[16 * H_SZ];      // h tile, row-major
  __shared__ __align__(16) __bf16 xls[2][16 * I_SZ];   // double-buffered x tile

  const int tid  = threadIdx.x;
  const int lane = tid & 31;
  const int wave = tid >> 5;                 // 0..15 : N-tile index
  const int dir  = blockIdx.x & 1;
  const int b0   = (blockIdx.x >> 1) * 16;

  const float* Wxh = dir ? Wxh_b : Wxh_f;
  const float* Whh = dir ? Whh_b : Whh_f;
  const float* bh  = dir ? bh_b  : bh_f;

  const int ncol = wave * 16 + (lane & 15);  // output column this lane owns
  const int ksel = (lane < 16) ? 0 : 8;      // B-frag K offset per half (table order)
  const float bias = bh[ncol];

  // ---- Load combined [Wxh;Whh] (K=384 x N=256) B-fragments into VGPRs ----
  v16bf bfrag[NKB];
  #pragma unroll
  for (int kb = 0; kb < NKB; ++kb) {
    v16bf f;
    #pragma unroll
    for (int i = 0; i < 16; ++i) {
      int k = kb * 32 + ksel + i + (i & 8);
      float w = (k < I_SZ) ? Wxh[k * H_SZ + ncol] : Whh[(k - I_SZ) * H_SZ + ncol];
      f[i] = (__bf16)w;
    }
    bfrag[kb] = f;
  }

  // ---- h0 = 0 ----
  {
    uint2 z; z.x = 0u; z.y = 0u;
    ((uint2*)hls)[tid]       = z;
    ((uint2*)hls)[tid + 512] = z;
  }

  // ---- x staging: each thread owns 4 consecutive floats of the 16x128 tile
  const int xe    = tid * 4;
  const int xrow  = xe >> 7;
  const int xcoli = xe & 127;
  const long long tstart = dir ? (T_LEN - 1) : 0;
  const long long tstep  = dir ? -(long long)I_SZ : (long long)I_SZ;
  const float* xptr = x + ((long long)(b0 + xrow) * T_LEN + tstart) * I_SZ + xcoli;

  float4 xv = *(const float4*)xptr;          // stage x(t=0)
  {
    uint2 p; p.x = pack2_bf16(xv.x, xv.y); p.y = pack2_bf16(xv.z, xv.w);
    ((uint2*)&xls[0][0])[tid] = p;
  }
  __syncthreads();

  const int arow   = lane & 15;              // A-matrix row this lane reads
  const int asel   = (lane < 16) ? 0 : 8;    // A-frag element offset per half
  const int rowoff = (lane < 16) ? 0 : 8;    // C/D layout: M = r (+8 hi lanes)

  float hlast[8];

  for (int t = 0; t < T_LEN; ++t) {
    // Issue next x tile load early; latency hidden under the WMMA work.
    if (t + 1 < T_LEN) {
      xptr += tstep;
      xv = *(const float4*)xptr;
      if (t + 2 < T_LEN) __builtin_prefetch(xptr + tstep, 0, 1); // global_prefetch_b8
    }

    const __bf16* xrb = &xls[t & 1][0] + arow * I_SZ;
    const __bf16* hrb = hls            + arow * H_SZ;

    // Two independent accumulation chains; chain A seeded with the bias.
    v8f accA, accB;
    #pragma unroll
    for (int r = 0; r < 8; ++r) { accA[r] = bias; accB[r] = 0.0f; }

    // Depth-2 software pipeline: keep 4 A-fragments in flight so each
    // ds_load pair issues ~4 WMMAs before its consumer (waits become
    // partial instead of s_wait_dscnt 0).
    v16bf a0 = load_a_frag(a_src(0, xrb, hrb), a_off(0, asel));
    v16bf a1 = load_a_frag(a_src(1, xrb, hrb), a_off(1, asel));
    v16bf a2 = load_a_frag(a_src(2, xrb, hrb), a_off(2, asel));
    v16bf a3 = load_a_frag(a_src(3, xrb, hrb), a_off(3, asel));
    #pragma unroll
    for (int kb = 0; kb < NKB; kb += 2) {
      v16bf c0 = a0, c1 = a1;
      a0 = a2; a1 = a3;
      if (kb + 4 < NKB) {
        a2 = load_a_frag(a_src(kb + 4, xrb, hrb), a_off(kb + 4, asel));
        a3 = load_a_frag(a_src(kb + 5, xrb, hrb), a_off(kb + 5, asel));
      }
      accA = __builtin_amdgcn_wmma_f32_16x16x32_bf16(
          false, c0, false, bfrag[kb], (short)0, accA, false, false);
      accB = __builtin_amdgcn_wmma_f32_16x16x32_bf16(
          false, c1, false, bfrag[kb + 1], (short)0, accB, false, false);
    }

    __syncthreads();   // all waves finished reading h(t) / x buffer

    // h(t+1) = tanh(x@Wxh + h@Whh + bh) ; write back bf16 for next step's A
    #pragma unroll
    for (int r = 0; r < 8; ++r) {
      float hv = fast_tanh(accA[r] + accB[r]);
      hlast[r] = hv;
      hls[(r + rowoff) * H_SZ + ncol] = (__bf16)hv;
    }

    if (t + 1 < T_LEN) {  // stage x(t+1) into the other buffer
      uint2 p; p.x = pack2_bf16(xv.x, xv.y); p.y = pack2_bf16(xv.z, xv.w);
      ((uint2*)&xls[(t + 1) & 1][0])[tid] = p;
    }
    __syncthreads();   // new h (and staged x) visible to all waves
  }

  // Final hidden state -> hcat[b, dir*H + n] in fp32
  #pragma unroll
  for (int r = 0; r < 8; ++r) {
    hcat[(long long)(b0 + r + rowoff) * (2 * H_SZ) + dir * H_SZ + ncol] = hlast[r];
  }
}

// ---------------------------------------------------------------------------
// Tiny output GEMM: out[128,128] = hcat[128,512] @ Why[512,128] + by (fp32)
// ---------------------------------------------------------------------------
__global__ void birnn_out(const float* __restrict__ hcat,
                          const float* __restrict__ Why,
                          const float* __restrict__ by,
                          float* __restrict__ out)
{
  int idx = blockIdx.x * blockDim.x + threadIdx.x;   // 0..16383
  int b = idx >> 7;
  int o = idx & 127;
  float s = by[o];
  const float* hr = hcat + (long long)b * (2 * H_SZ);
  #pragma unroll 8
  for (int k = 0; k < 2 * H_SZ; ++k)
    s = fmaf(hr[k], Why[k * 128 + o], s);
  out[idx] = s;
}

extern "C" void kernel_launch(void* const* d_in, const int* in_sizes, int n_in,
                              void* d_out, int out_size, void* d_ws, size_t ws_size,
                              hipStream_t stream) {
  (void)in_sizes; (void)n_in; (void)out_size; (void)ws_size;
  const float* x     = (const float*)d_in[0];
  const float* Wxh_f = (const float*)d_in[1];
  const float* Whh_f = (const float*)d_in[2];
  const float* bh_f  = (const float*)d_in[3];
  const float* Wxh_b = (const float*)d_in[4];
  const float* Whh_b = (const float*)d_in[5];
  const float* bh_b  = (const float*)d_in[6];
  const float* Why   = (const float*)d_in[7];
  const float* by    = (const float*)d_in[8];
  float* hcat = (float*)d_ws;                 // 128*512*4 = 256 KB scratch
  float* out  = (float*)d_out;

  // 16 persistent workgroups: 8 batch tiles x 2 directions
  birnn_scan<<<dim3(16), dim3(512), 0, stream>>>(
      x, Wxh_f, Whh_f, bh_f, Wxh_b, Whh_b, bh_b, hcat);
  birnn_out<<<dim3(64), dim3(256), 0, stream>>>(hcat, Why, by, out);
}
